// ImageRetResNet_31636729102547
// MI455X (gfx1250) — matrix operand order repaired and checked by reading, not compiled, not run on hardware
//
#include <hip/hip_runtime.h>

typedef __attribute__((ext_vector_type(16))) __bf16 v16bf;
typedef __attribute__((ext_vector_type(8)))  __bf16 v8bf;
typedef __attribute__((ext_vector_type(8)))  float  v8f;
typedef __attribute__((ext_vector_type(4)))  unsigned int u32x4;
typedef __attribute__((ext_vector_type(8)))  int i32x8;
typedef __attribute__((ext_vector_type(4)))  int i32x4;

#define EPSV 1e-5f

// ==================== batched WMMA GEMM (bf16 operands, f32 acc) ====================
// C = A * B^T with A: [M,K] bf16 row-major (lda), B: [N,K] bf16 row-major (ldb)
// (every B operand is pre-transposed by its producer so K is contiguous).
// M % 32 == 0 and K % 32 == 0 guaranteed by callers; N may be ragged (TDM
// zero-fills rows past tensor_dim1). Tiles move global->LDS with the Tensor
// Data Mover (tensor_load_to_lds, TENSORcnt), double-buffered, one barrier
// per K-step, one v_wmma_f32_16x16x32_bf16 per wave per K-step.
constexpr int TM = 32, TN = 64, TK = 32;
constexpr int TKP = TK + 8;   // LDS row stride in halves: 80B = 64B data + 16B TDM pad

// Issue one TDM 2D tile load: rows x 32 bf16 from gptr (row stride ld elements)
// into LDS at byte offset ldsaddr, padding each 64B row to 80B.
// krem/nrem: remaining tensor extent from the tile origin (OOB rows -> zeros).
__device__ __forceinline__ void tdm_load_tile(const __bf16* gptr, unsigned ldsaddr,
                                              int ld, int krem, int nrem, int rows)
{
  unsigned long long ga = (unsigned long long)gptr;
  u32x4 g0;
  g0[0] = 1u;                                   // count=1, user descriptor
  g0[1] = ldsaddr;                              // lds_addr (bytes)
  g0[2] = (unsigned)ga;                         // global_addr[31:0]
  g0[3] = (unsigned)((ga >> 32) & 0x01FFFFFFu) | (2u << 30);  // addr[56:32] | type=2
  i32x8 g1;
  // data_size=1 (2B) | pad_enable | pad_interval=3 (64B) | pad_amount=3 (16B)
  g1[0] = (int)((1u << 16) | (1u << 20) | (3u << 22) | (3u << 25));
  g1[1] = (int)(((unsigned)krem & 0xFFFFu) << 16);                         // t_dim0 lo
  g1[2] = (int)((((unsigned)krem >> 16) & 0xFFFFu) |
                (((unsigned)nrem & 0xFFFFu) << 16));                       // t_dim0 hi | t_dim1 lo
  g1[3] = (int)((((unsigned)nrem >> 16) & 0xFFFFu) | (32u << 16));         // t_dim1 hi | tile_dim0=32
  g1[4] = rows;                                                            // tile_dim1
  g1[5] = (int)(unsigned)ld;                                               // t_dim0_stride lo
  g1[6] = 0;                                                               // stride hi | t_dim1_stride
  g1[7] = 0;
  i32x4 gz4 = {0, 0, 0, 0};
  i32x8 gz8 = {0, 0, 0, 0, 0, 0, 0, 0};
  __builtin_amdgcn_tensor_load_to_lds(g0, g1, gz4, gz4, gz8, 0);
}

__global__ __launch_bounds__(256)
void k_gemm(const __bf16* __restrict__ A, const __bf16* __restrict__ B,
            float* __restrict__ C,
            int M, int N, int K, int lda, int ldb, int ldc,
            long long sA1, long long sB1, long long sC1,
            long long sA2, long long sB2, long long sC2,
            int nB2)
{
  const int b1 = blockIdx.z / nB2;
  const int b2 = blockIdx.z % nB2;
  A += b1 * sA1 + b2 * sA2;
  B += b1 * sB1 + b2 * sB2;
  C += b1 * sC1 + b2 * sC2;

  const int m0 = blockIdx.y * TM;
  const int n0 = blockIdx.x * TN;

  __shared__ __bf16 As[2][TM][TKP];
  __shared__ __bf16 Bs[2][TN][TKP];

  const int tid  = threadIdx.x;
  const int lane = tid & 31;
  const int wave = tid >> 5;
  const int wm = (wave & 1) * 16;
  const int wn = (wave >> 1) * 16;

  const unsigned ldsA0 = (unsigned)(unsigned long long)(const void*)&As[0][0][0];
  const unsigned ldsA1 = (unsigned)(unsigned long long)(const void*)&As[1][0][0];
  const unsigned ldsB0 = (unsigned)(unsigned long long)(const void*)&Bs[0][0][0];
  const unsigned ldsB1 = (unsigned)(unsigned long long)(const void*)&Bs[1][0][0];

  const int niter = K / TK;
  // prologue: DMA K-tile 0 into buffer 0 (wave 0 owns the descriptors)
  if (wave == 0) {
    tdm_load_tile(A + (long long)m0 * lda, ldsA0, lda, K, M - m0, TM);
    tdm_load_tile(B + (long long)n0 * ldb, ldsB0, ldb, K, N - n0, TN);
  }

  v8f acc = {0.f,0.f,0.f,0.f,0.f,0.f,0.f,0.f};

  for (int it = 0; it < niter; ++it) {
    const int buf = it & 1;
    if (wave == 0) __builtin_amdgcn_s_wait_tensorcnt((short)0);
    __syncthreads();
    // overlap: DMA next K-tile into the other buffer while we compute this one
    if (wave == 0 && it + 1 < niter) {
      const int k1 = (it + 1) * TK;
      tdm_load_tile(A + (long long)m0 * lda + k1, buf ? ldsA0 : ldsA1,
                    lda, K - k1, M - m0, TM);
      tdm_load_tile(B + (long long)n0 * ldb + k1, buf ? ldsB0 : ldsB1,
                    ldb, K - k1, N - n0, TN);
    }

    // fragments: two contiguous 8-half runs each -> ds_load_b128 pairs
    const int mr  = wm + (lane & 15);
    const int kh2 = (lane >> 4) << 3;    // 0 or 8
    v8bf a0 = *(const v8bf*)&As[buf][mr][kh2];
    v8bf a1 = *(const v8bf*)&As[buf][mr][kh2 + 16];
    v16bf af = __builtin_shufflevector(a0, a1, 0,1,2,3,4,5,6,7,8,9,10,11,12,13,14,15);

    const int nc = wn + (lane & 15);
    const int kb = (lane >> 4) << 3;     // 0 or 8
    v8bf b0 = *(const v8bf*)&Bs[buf][nc][kb];
    v8bf b1 = *(const v8bf*)&Bs[buf][nc][kb + 16];
    v16bf bfv = __builtin_shufflevector(b0, b1, 0,1,2,3,4,5,6,7,8,9,10,11,12,13,14,15);

    acc = __builtin_amdgcn_wmma_f32_16x16x32_bf16(
        false, af, false, bfv, (short)0, acc, false, false);
  }

  // C/D layout: VGPR r -> row r (lanes 0-15) / 8+r (lanes 16-31), col = lane&15
  const int gn = n0 + wn + (lane & 15);
  if (gn < N) {
    const int mb = m0 + wm + ((lane >> 4) << 3);
#pragma unroll
    for (int r = 0; r < 8; ++r)
      C[(long long)(mb + r) * ldc + gn] = acc[r];
  }
}

// ==================== data-movement / elementwise kernels ====================

__global__ void k_tonhwc(const float* __restrict__ in, float* __restrict__ out,
                         int N, int C, int H, int W, long long total)
{
  long long idx = (long long)blockIdx.x * blockDim.x + threadIdx.x;
  if (idx >= total) return;
  int w = idx % W; long long t = idx / W;
  int h = t % H; t /= H;
  int c = t % C; long long n = t / C;
  out[(((long long)n * H + h) * W + w) * C + c] = in[idx];
}

// im2col producing bf16 patch rows [rows, Kp], j = (kh*KW + kw)*C + ci
__global__ void k_im2col(const float* __restrict__ x, __bf16* __restrict__ cols,
                         int H, int W, int C, int KH, int KW,
                         int stride, int pad, int OH, int OW, int Kp, long long total)
{
  long long idx = (long long)blockIdx.x * blockDim.x + threadIdx.x;
  if (idx >= total) return;
  int j = idx % Kp; long long r = idx / Kp;
  int ow = r % OW; long long t = r / OW;
  int oh = t % OH; long long n = t / OH;
  float v = 0.f;
  if (j < KH * KW * C) {
    int c = j % C; int kw = (j / C) % KW; int kh = j / (C * KW);
    int ih = oh * stride - pad + kh, iw = ow * stride - pad + kw;
    if (ih >= 0 && ih < H && iw >= 0 && iw < W)
      v = x[(((long long)n * H + ih) * W + iw) * C + c];
  }
  cols[idx] = (__bf16)v;
}

// OIHW weights -> bf16 [CO, Kp] (K contiguous)
__global__ void k_wconv(const float* __restrict__ w, __bf16* __restrict__ o,
                        int CO, int CI, int KH, int KW, int Kp, long long total)
{
  long long idx = (long long)blockIdx.x * blockDim.x + threadIdx.x;
  if (idx >= total) return;
  int j = idx % Kp; int co = (int)(idx / Kp);
  float v = 0.f;
  if (j < KH * KW * CI) {
    int c = j % CI; int kw = (j / CI) % KW; int kh = j / (CI * KW);
    v = w[(((long long)co * CI + c) * KH + kh) * KW + kw];
  }
  o[idx] = (__bf16)v;
}

// [H, d, hd] -> bf16 [H*hd, d] (transposed projection weights, K=d contiguous)
__global__ void k_wret(const float* __restrict__ w, __bf16* __restrict__ o,
                       int Hh, int d, int hd, long long total)
{
  long long idx = (long long)blockIdx.x * blockDim.x + threadIdx.x;
  if (idx >= total) return;
  int dd = idx % d; int n = (int)(idx / d);   // n = h*hd + e
  int h = n / hd, e = n % hd;
  o[idx] = (__bf16)w[((long long)h * d + dd) * hd + e];
}

// f32 [R, Ccol] -> bf16 [Ccol, R] (transpose; for WG/WO)
__global__ void k_castT(const float* __restrict__ in, __bf16* __restrict__ o,
                        int R, int Ccol, long long total)
{
  long long idx = (long long)blockIdx.x * blockDim.x + threadIdx.x;
  if (idx >= total) return;
  int k = idx % R; int n = (int)(idx / R);
  o[idx] = (__bf16)in[(long long)k * Ccol + n];
}

__global__ void k_cast(const float* __restrict__ in, __bf16* __restrict__ o,
                       long long total)
{
  long long idx = (long long)blockIdx.x * blockDim.x + threadIdx.x;
  if (idx >= total) return;
  o[idx] = (__bf16)in[idx];
}

// V f32 [B, L, H*hd] -> Vt bf16 [B, H, hd, L] (K=L contiguous per (b,h))
__global__ void k_vt(const float* __restrict__ V, __bf16* __restrict__ o,
                     int L, int Hh, int hd, long long total)
{
  long long idx = (long long)blockIdx.x * blockDim.x + threadIdx.x;
  if (idx >= total) return;
  int l = idx % L; long long t = idx / L;
  int e = t % hd; t /= hd;
  int h = t % Hh; long long b = t / Hh;
  o[idx] = (__bf16)V[((b * L + l) * (long long)(Hh * hd)) + (long long)h * hd + e];
}

__global__ void k_bnrelu(float* __restrict__ y, const float* __restrict__ b,
                         const float* __restrict__ bw, const float* __restrict__ bb,
                         const float* __restrict__ bm, const float* __restrict__ bv,
                         int C, long long total)
{
  long long idx = (long long)blockIdx.x * blockDim.x + threadIdx.x;
  if (idx >= total) return;
  int c = idx % C;
  float v = y[idx] + b[c];
  v = (v - bm[c]) * rsqrtf(bv[c] + EPSV) * bw[c] + bb[c];
  y[idx] = v > 0.f ? v : 0.f;
}

// xpos rotary: f32 [B,L,H*hd] -> bf16 [B,L,H*hd]; down=1 -> 1/scale
__global__ void k_xpos(const float* __restrict__ x, __bf16* __restrict__ o,
                       int L, int Hh, int hd, int down, long long total)
{
  long long idx = (long long)blockIdx.x * blockDim.x + threadIdx.x;
  if (idx >= total) return;
  int half = hd >> 1;
  int j = idx % half; long long t = idx / half;
  int h = t % Hh; t /= Hh;
  int l = t % L; long long bb = t / L;
  float s = ((float)(j * 2) + 0.4f * (float)hd) / (1.4f * (float)hd);
  float pos = (float)l;
  float scale = __powf(s, pos / 512.0f);
  if (down) scale = 1.0f / scale;
  float invf = __powf(10000.0f, -(float)j / (float)half);
  float ang = pos * invf;
  float sn = __sinf(ang) * scale, cs = __cosf(ang) * scale;
  long long base = ((bb * L + l) * (long long)(Hh * hd)) + (long long)h * hd + 2 * j;
  float x1 = x[base], x2 = x[base + 1];
  o[base]     = (__bf16)(x1 * cs - x2 * sn);
  o[base + 1] = (__bf16)(x2 * cs + x1 * sn);
}

// retention decay mask: f32 A [B,H,L,L] -> bf16 (masked/scaled)
__global__ void k_decay(const float* __restrict__ Aa, __bf16* __restrict__ o,
                        int Hh, int Lq, long long total)
{
  long long idx = (long long)blockIdx.x * blockDim.x + threadIdx.x;
  if (idx >= total) return;
  int mm = idx % Lq; long long t = idx / Lq;
  int nn = t % Lq; t /= Lq;
  int h = t % Hh;
  const float l0 = -3.46573590f;   // log(1/32)
  const float l1 = -6.23832463f;   // log(1/512)
  float gamma = 1.0f - __expf(l0 + (l1 - l0) * (float)h / (float)(Hh - 1));
  float v = (nn >= mm) ? Aa[idx] * __powf(gamma, (float)(nn - mm)) : 0.f;
  o[idx] = (__bf16)v;
}

// per-head group norm over [BL, H*hd] then affine with full-width gn_w/gn_b
__global__ void k_groupnorm(const float* __restrict__ Y, float* __restrict__ Yn,
                            const float* __restrict__ gw, const float* __restrict__ gb,
                            int Hh, int hd, long long total /* = BL*Hh */)
{
  long long idx = (long long)blockIdx.x * blockDim.x + threadIdx.x;
  if (idx >= total) return;
  int h = idx % Hh; long long row = idx / Hh;
  const float* y = Y + row * (long long)(Hh * hd) + (long long)h * hd;
  float mu = 0.f;
  for (int e = 0; e < hd; ++e) mu += y[e];
  mu /= (float)hd;
  float var = 0.f;
  for (int e = 0; e < hd; ++e) { float dv = y[e] - mu; var += dv * dv; }
  var /= (float)hd;
  float rs = rsqrtf(var + EPSV);
  float* o = Yn + row * (long long)(Hh * hd) + (long long)h * hd;
  for (int e = 0; e < hd; ++e)
    o[e] = (y[e] - mu) * rs * gw[h * hd + e] + gb[h * hd + e];
}

// z = silu(g) * yn, bf16 out (A operand of the WO GEMM)
__global__ void k_silu_mul(const float* __restrict__ g, const float* __restrict__ yn,
                           __bf16* __restrict__ z, long long total)
{
  long long idx = (long long)blockIdx.x * blockDim.x + threadIdx.x;
  if (idx >= total) return;
  float gv = g[idx];
  z[idx] = (__bf16)((gv / (1.0f + __expf(-gv))) * yn[idx]);
}

__global__ void k_add(float* __restrict__ dst, const float* __restrict__ a,
                      const float* __restrict__ b, long long total)
{
  long long idx = (long long)blockIdx.x * blockDim.x + threadIdx.x;
  if (idx >= total) return;
  dst[idx] = a[idx] + b[idx];
}

__global__ void k_mean(const float* __restrict__ x, float* __restrict__ o,
                       int L, int C, long long total /* = B*C */)
{
  long long idx = (long long)blockIdx.x * blockDim.x + threadIdx.x;
  if (idx >= total) return;
  int c = idx % C; long long b = idx / C;
  float s = 0.f;
  for (int l = 0; l < L; ++l) s += x[((long long)b * L + l) * C + c];
  o[idx] = s / (float)L;
}

__global__ void k_cls(const float* __restrict__ f, const float* __restrict__ w,
                      const float* __restrict__ bias, float* __restrict__ o,
                      int C, int O, long long total /* = B*O */)
{
  long long idx = (long long)blockIdx.x * blockDim.x + threadIdx.x;
  if (idx >= total) return;
  int j = idx % O; long long b = idx / O;
  float s = bias[j];
  for (int c = 0; c < C; ++c) s += f[b * C + c] * w[(long long)c * O + j];
  o[idx] = s;
}

// ==================== host-side orchestration ====================

struct ConvP { const float *b, *bn_b, *bn_m, *bn_v, *bn_w, *w; };
struct RetP  { const float *WG, *WK, *WO, *WQ, *WV, *gn_b, *gn_w; };

static inline ConvP getConv(void* const* din, int base) {
  return { (const float*)din[base + 0], (const float*)din[base + 1],
           (const float*)din[base + 2], (const float*)din[base + 3],
           (const float*)din[base + 4], (const float*)din[base + 5] };
}
static inline RetP getRet(void* const* din, int base) {
  return { (const float*)din[base + 0], (const float*)din[base + 1],
           (const float*)din[base + 2], (const float*)din[base + 3],
           (const float*)din[base + 4], (const float*)din[base + 5],
           (const float*)din[base + 6] };
}

static inline int g1(long long n) { return (int)((n + 255) / 256); }

static void gemm(hipStream_t st, const __bf16* A, const __bf16* B, float* C,
                 int M, int N, int K, int lda, int ldb, int ldc,
                 int B1, long long sA1, long long sB1, long long sC1,
                 int B2, long long sA2, long long sB2, long long sC2)
{
  dim3 grid((N + TN - 1) / TN, (M + TM - 1) / TM, B1 * B2);
  k_gemm<<<grid, 256, 0, st>>>(A, B, C, M, N, K, lda, ldb, ldc,
                               sA1, sB1, sC1, sA2, sB2, sC2, B2);
}

// NHWC conv + bias + BN + ReLU via bf16 im2col + TDM/WMMA GEMM
static void conv_bnrelu(hipStream_t st, const float* xin, float* out, const ConvP& p,
                        int N, int H, int W, int CI, int CO,
                        int KH, int KW, int stride, int pad,
                        __bf16* cols, __bf16* wbuf)
{
  int OH = (H + 2 * pad - KH) / stride + 1;
  int OW = (W + 2 * pad - KW) / stride + 1;
  int Kfull = KH * KW * CI;
  int Kp = (Kfull + 31) & ~31;
  long long rows = (long long)N * OH * OW;
  long long wtot = (long long)Kp * CO;
  k_wconv<<<g1(wtot), 256, 0, st>>>(p.w, wbuf, CO, CI, KH, KW, Kp, wtot);
  long long ctot = rows * Kp;
  k_im2col<<<g1(ctot), 256, 0, st>>>(xin, cols, H, W, CI, KH, KW, stride, pad, OH, OW, Kp, ctot);
  gemm(st, cols, wbuf, out, (int)rows, CO, Kp, Kp, Kp, CO,
       1, 0, 0, 0, 1, 0, 0, 0);
  k_bnrelu<<<g1(rows * CO), 256, 0, st>>>(out, p.b, p.bn_w, p.bn_b, p.bn_m, p.bn_v,
                                          CO, rows * CO);
}

// parallel retention: x f32 [B,L,d] -> yout f32 [B,L,d]
static void msr_parallel(hipStream_t st, const float* x, float* yout, const RetP& p,
                         int B, int L, int d,
                         __bf16* xb, __bf16* wqT, __bf16* wkT, __bf16* wvT,
                         __bf16* wgT, __bf16* woT,
                         float* Q, float* Kt, float* V,
                         __bf16* Qb, __bf16* Ktb, __bf16* Vtb,
                         float* Ab, __bf16* Abb,
                         float* Y, float* Yn, float* Gf, __bf16* Gb)
{
  const int Hh = 4, hd = d / Hh;
  const long long BL = (long long)B * L;
  const long long dd = (long long)d * d;
  k_cast <<<g1(BL * d), 256, 0, st>>>(x, xb, BL * d);
  k_wret <<<g1(dd), 256, 0, st>>>(p.WQ, wqT, Hh, d, hd, dd);
  k_wret <<<g1(dd), 256, 0, st>>>(p.WK, wkT, Hh, d, hd, dd);
  k_wret <<<g1(dd), 256, 0, st>>>(p.WV, wvT, Hh, d, hd, dd);
  k_castT<<<g1(dd), 256, 0, st>>>(p.WG, wgT, d, d, dd);
  k_castT<<<g1(dd), 256, 0, st>>>(p.WO, woT, d, d, dd);
  // projections: [B*L, d] @ [d, d]  (weights stored transposed [d_out, d_in])
  gemm(st, xb, wqT, Q,  (int)BL, d, d, d, d, d, 1,0,0,0, 1,0,0,0);
  gemm(st, xb, wkT, Kt, (int)BL, d, d, d, d, d, 1,0,0,0, 1,0,0,0);
  gemm(st, xb, wvT, V,  (int)BL, d, d, d, d, d, 1,0,0,0, 1,0,0,0);
  k_xpos<<<g1(BL * Hh * (hd / 2)), 256, 0, st>>>(Q,  Qb,  L, Hh, hd, 0, BL * Hh * (hd / 2));
  k_xpos<<<g1(BL * Hh * (hd / 2)), 256, 0, st>>>(Kt, Ktb, L, Hh, hd, 1, BL * Hh * (hd / 2));
  k_vt  <<<g1(BL * d), 256, 0, st>>>(V, Vtb, L, Hh, hd, BL * d);
  // A = Q K^T per (b,h): [L,hd] x [L,hd]^T -> [L,L], stored f32 [B,H,L,L]
  gemm(st, Qb, Ktb, Ab, L, L, hd, d, d, L,
       B,  (long long)L * d, (long long)L * d, (long long)Hh * L * L,
       Hh, hd,               hd,               (long long)L * L);
  long long atot = (long long)B * Hh * L * L;
  k_decay<<<g1(atot), 256, 0, st>>>(Ab, Abb, Hh, L, atot);
  // Y = A V per (b,h): [L,L] x [hd,L]^T -> written into f32 [B,L,H*hd]
  gemm(st, Abb, Vtb, Y, L, hd, L, L, L, d,
       B,  (long long)Hh * L * L, (long long)Hh * hd * L, (long long)L * d,
       Hh, (long long)L * L,      (long long)hd * L,      hd);
  // gate_out
  k_groupnorm<<<g1(BL * Hh), 256, 0, st>>>(Y, Yn, p.gn_w, p.gn_b, Hh, hd, BL * Hh);
  gemm(st, xb, wgT, Gf, (int)BL, d, d, d, d, d, 1,0,0,0, 1,0,0,0);
  k_silu_mul<<<g1(BL * d), 256, 0, st>>>(Gf, Yn, Gb, BL * d);
  gemm(st, Gb, woT, yout, (int)BL, d, d, d, d, d, 1,0,0,0, 1,0,0,0);
}

extern "C" void kernel_launch(void* const* d_in, const int* in_sizes, int n_in,
                              void* d_out, int out_size, void* d_ws, size_t ws_size,
                              hipStream_t stream)
{
  (void)in_sizes; (void)n_in; (void)out_size; (void)ws_size;
  // ---- input map (jax tree flatten: dict keys sorted alphabetically) ----
  // 0: cls.b  1: cls.w
  // dense[i] @ 2+6i, ret[i] @ 20+7i, s2i[i] @ 62+6i, stem @ 86, x @ 92
  const float* cls_b = (const float*)d_in[0];
  const float* cls_w = (const float*)d_in[1];
  ConvP dense0 = getConv(d_in, 2);
  ConvP dense1 = getConv(d_in, 8);
  ConvP dense2 = getConv(d_in, 14);
  RetP  ret4   = getRet(d_in, 48);
  RetP  ret5   = getRet(d_in, 55);
  ConvP s2i3   = getConv(d_in, 80);
  ConvP stem   = getConv(d_in, 86);
  const float* x = (const float*)d_in[92];

  // ---- workspace partition (byte bump allocator) ----
  char* base = (char*)d_ws;
  size_t off = 0;
  auto alloc = [&](size_t bytes) {
    void* p = base + off; off += (bytes + 255) & ~(size_t)255; return p;
  };
  float*  xh    = (float*)alloc(32ull * 256 * 256 * 3 * 4);   // NHWC input
  __bf16* cols  = (__bf16*)alloc(131072ull * 320 * 2);        // bf16 im2col (max: stem)
  __bf16* wbuf  = (__bf16*)alloc(2304ull * 512 * 2);          // bf16 conv weights (max: s2i3)
  float*  f1    = (float*)alloc(32ull * 64 * 64 * 32 * 4);
  float*  f2    = (float*)alloc(32ull * 32 * 32 * 64 * 4);
  float*  f3    = (float*)alloc(32ull * 16 * 16 * 128 * 4);
  float*  f4    = (float*)alloc(32ull * 8 * 8 * 256 * 4);     // x4: [32, 64, 256]
  const size_t SEQ = 32ull * 64 * 256;                        // B*L*d elements
  __bf16* xb    = (__bf16*)alloc(SEQ * 2);
  __bf16* wqT   = (__bf16*)alloc(256ull * 256 * 2);
  __bf16* wkT   = (__bf16*)alloc(256ull * 256 * 2);
  __bf16* wvT   = (__bf16*)alloc(256ull * 256 * 2);
  __bf16* wgT   = (__bf16*)alloc(256ull * 256 * 2);
  __bf16* woT   = (__bf16*)alloc(256ull * 256 * 2);
  float*  Q     = (float*)alloc(SEQ * 4);
  float*  Kt    = (float*)alloc(SEQ * 4);
  float*  V     = (float*)alloc(SEQ * 4);
  __bf16* Qb    = (__bf16*)alloc(SEQ * 2);
  __bf16* Ktb   = (__bf16*)alloc(SEQ * 2);
  __bf16* Vtb   = (__bf16*)alloc(SEQ * 2);
  float*  Ab    = (float*)alloc(32ull * 4 * 64 * 64 * 4);
  __bf16* Abb   = (__bf16*)alloc(32ull * 4 * 64 * 64 * 2);
  float*  Y     = (float*)alloc(SEQ * 4);
  float*  Yn    = (float*)alloc(SEQ * 4);
  float*  Gf    = (float*)alloc(SEQ * 4);
  __bf16* Gb    = (__bf16*)alloc(SEQ * 2);
  float*  y1    = (float*)alloc(SEQ * 4);
  float*  y2    = (float*)alloc(SEQ * 4);
  float*  sumb  = (float*)alloc(SEQ * 4);
  float*  seq5  = (float*)alloc(32ull * 16 * 512 * 4);
  float*  meanb = (float*)alloc(32ull * 512 * 4);

  const int B = 32;

  // Live dataflow only (y is reassigned from x3 and x4 in the reference, so
  // ret[0..3], s2i[0..2] and the chunkwise path are dead code):
  long long xtot = 32ll * 3 * 256 * 256;
  k_tonhwc<<<g1(xtot), 256, 0, stream>>>(x, xh, 32, 3, 256, 256, xtot);

  conv_bnrelu(stream, xh, f1, stem,   B, 256, 256,   3,  32, 10, 10, 4, 3, cols, wbuf); // 64x64x32
  conv_bnrelu(stream, f1, f2, dense0, B,  64,  64,  32,  64,  3,  3, 2, 1, cols, wbuf); // 32x32x64
  conv_bnrelu(stream, f2, f3, dense1, B,  32,  32,  64, 128,  3,  3, 2, 1, cols, wbuf); // 16x16x128
  conv_bnrelu(stream, f3, f4, dense2, B,  16,  16, 128, 256,  3,  3, 2, 1, cols, wbuf); // 8x8x256

  // x4 = f4 viewed as [32, 64, 256]
  msr_parallel(stream, f4, y1, ret4, B, 64, 256, xb, wqT, wkT, wvT, wgT, woT,
               Q, Kt, V, Qb, Ktb, Vtb, Ab, Abb, Y, Yn, Gf, Gb);
  msr_parallel(stream, y1, y2, ret5, B, 64, 256, xb, wqT, wkT, wvT, wgT, woT,
               Q, Kt, V, Qb, Ktb, Vtb, Ab, Abb, Y, Yn, Gf, Gb);

  long long stot = 32ll * 64 * 256;
  k_add<<<g1(stot), 256, 0, stream>>>(sumb, f4, y2, stot);

  // s2i2s with s2i[3]: [32,8,8,256] -> [32,4,4,512] -> seq [32,16,512]
  conv_bnrelu(stream, sumb, seq5, s2i3, B, 8, 8, 256, 512, 3, 3, 2, 1, cols, wbuf);

  k_mean<<<g1(32ll * 512), 256, 0, stream>>>(seq5, meanb, 16, 512, 32ll * 512);
  k_cls<<<g1(32ll * 10), 256, 0, stream>>>(meanb, cls_w, cls_b, (float*)d_out,
                                           512, 10, 32ll * 10);
}
